// LucasKANLayer_5299989643403
// MI455X (gfx1250) — matrix operand order repaired and checked by reading, not compiled, not run on hardware
//
#include <hip/hip_runtime.h>
#include <hip/hip_bf16.h>

typedef __attribute__((ext_vector_type(16))) __bf16 v16bf;
typedef __attribute__((ext_vector_type(8)))  float  v8f;

#define MROWS 8192
#define IDIMS 1024
#define NOUT  1024
#define NDEG  8            // DEGREE+1
#define KDIM  (IDIMS*NDEG) // 8192

#define BM 128
#define BN 128
#define KC 64              // 8 input dims * 8 degrees per K-chunk
#define LDA 72             // padded LDS row stride (conflict-free, 16B aligned)
#define LDB 72

// ---------------- Pass 1a: T = bf16(tanh(x)), vectorized x4 ----------------
__global__ __launch_bounds__(256) void tanh_bf16_kernel(const float* __restrict__ x,
                                                        __bf16* __restrict__ T) {
    int idx = blockIdx.x * 256 + threadIdx.x;        // over n/4 groups
    const float4* x4 = (const float4*)x;
    float4 v = x4[idx];
    union { __bf16 h[4]; uint2 u2; } o;
    o.h[0] = (__bf16)tanhf(v.x);
    o.h[1] = (__bf16)tanhf(v.y);
    o.h[2] = (__bf16)tanhf(v.z);
    o.h[3] = (__bf16)tanhf(v.w);
    ((uint2*)T)[idx] = o.u2;
}

// -- Pass 1b: Bt[n][k=i*8+d] = bf16(coeffs[i][n][d])  (pre-transposed, coalesced writes) --
__global__ __launch_bounds__(256) void packBt_kernel(const float* __restrict__ coeffs,
                                                     __bf16* __restrict__ Bt) {
    int idx = blockIdx.x * 256 + threadIdx.x;        // over NOUT*KDIM (n-major)
    int n = idx >> 13;                               // 0..1023
    int k = idx & 8191;                              // 0..8191
    int i = k >> 3, d = k & 7;
    Bt[idx] = (__bf16)coeffs[((i << 10) + n) * NDEG + d];
}

// ---------------- Pass 2: WMMA GEMM with on-the-fly Lucas basis ----------------
__global__ __launch_bounds__(256) void lucas_wmma_gemm(const __bf16* __restrict__ T,
                                                       const __bf16* __restrict__ Bt,
                                                       float* __restrict__ C) {
    __shared__ __bf16 lA[BM * LDA];   // [m][k] row-major, padded
    __shared__ __bf16 lB[BN * LDB];   // [n][k] row-major (B pre-transposed), padded

    const int tid  = threadIdx.x;
    const int m0   = blockIdx.y * BM;
    const int n0   = blockIdx.x * BN;
    const int w    = tid >> 5;
    const int lane = tid & 31;
    const int wm   = (w & 1) * 64;   // wave M offset inside tile
    const int wn   = (w >> 1) * 32;  // wave N offset inside tile
    const int half = lane >> 4;
    const int r    = lane & 15;

    const int ml    = tid >> 1;      // A-gen: row 0..127
    const int ihalf = tid & 1;       // A-gen: dim group 0..1 (4 dims each)
    const int nl    = tid >> 1;      // B-fill: n-row 0..127
    const int part  = tid & 1;       // B-fill: 64B half of the 128B row

    v8f acc[4][2] = {};

    for (int i0 = 0; i0 < IDIMS; i0 += KC / NDEG) {   // 8 input dims per chunk
        __syncthreads();

        // ---- A tile: 128 rows x 64 (8 dims * 8 degrees) Lucas basis ----
        union { __bf16 h[4]; uint2 u; } tv;
        tv.u = *(const uint2*)&T[(size_t)(m0 + ml) * IDIMS + i0 + ihalf * 4];
        #pragma unroll
        for (int c = 0; c < 4; ++c) {
            float t = (float)tv.h[c];
            union { __bf16 h[8]; uint4 u4; } v;
            float l0 = 2.0f, l1 = t;
            v.h[0] = (__bf16)l0;
            v.h[1] = (__bf16)l1;
            #pragma unroll
            for (int d = 2; d < NDEG; ++d) {
                float l2 = t * l1 + l0;
                v.h[d] = (__bf16)l2;
                l0 = l1; l1 = l2;
            }
            *(uint4*)&lA[ml * LDA + (ihalf * 4 + c) * NDEG] = v.u4;   // 16B aligned
        }

        // ---- B tile: 128 n-rows x 64 k, contiguous global -> b128 both sides ----
        {
            const uint4* g = (const uint4*)&Bt[(size_t)(n0 + nl) * KDIM + i0 * NDEG + part * 32];
            uint4*       s = (uint4*)&lB[nl * LDB + part * 32];
            #pragma unroll
            for (int c = 0; c < 4; ++c) s[c] = g[c];
        }
        __syncthreads();

        // ---- two K=32 sub-steps: 16 WMMAs per LDS fill ----
        #pragma unroll
        for (int kk = 0; kk < KC; kk += 32) {
            union { v16bf v; unsigned u[8]; } af[4], bfr[2];
            #pragma unroll
            for (int a = 0; a < 4; ++a) {
                const __bf16* base = &lA[(wm + a * 16 + r) * LDA + kk];
                #pragma unroll
                for (int j = 0; j < 8; ++j) {
                    int k0 = half * 16 + 2 * j;
                    af[a].u[j] = *(const unsigned*)&base[k0];
                }
            }
            #pragma unroll
            for (int b = 0; b < 2; ++b) {
                const __bf16* base = &lB[(wn + b * 16 + r) * LDB + kk + half * 16];
                #pragma unroll
                for (int j = 0; j < 8; ++j)
                    bfr[b].u[j] = *(const unsigned*)&base[2 * j];
            }
            #pragma unroll
            for (int a = 0; a < 4; ++a)
                #pragma unroll
                for (int b = 0; b < 2; ++b)
                    acc[a][b] = __builtin_amdgcn_wmma_f32_16x16x32_bf16(
                        false, af[a].v, false, bfr[b].v, (short)0, acc[a][b], false, false);
        }
    }

    // ---- write back: C/D layout N=lane%16, M=vgpr + 8*half ----
    #pragma unroll
    for (int a = 0; a < 4; ++a)
        #pragma unroll
        for (int b = 0; b < 2; ++b) {
            int ncol  = n0 + wn + b * 16 + r;
            int mbase = m0 + wm + a * 16 + half * 8;
            #pragma unroll
            for (int j = 0; j < 8; ++j)
                C[(size_t)(mbase + j) * NOUT + ncol] = acc[a][b][j];
        }
}

extern "C" void kernel_launch(void* const* d_in, const int* in_sizes, int n_in,
                              void* d_out, int out_size, void* d_ws, size_t ws_size,
                              hipStream_t stream) {
    const float* x      = (const float*)d_in[0];   // [8192,1024] f32
    const float* coeffs = (const float*)d_in[1];   // [1024,1024,8] f32
    float* out          = (float*)d_out;           // [8192,1024] f32

    // workspace layout: [0,16MB) tanh bf16, [16MB,32MB) transposed B bf16
    __bf16* T  = (__bf16*)d_ws;
    __bf16* Bt = (__bf16*)((char*)d_ws + (size_t)MROWS * IDIMS * sizeof(__bf16));

    tanh_bf16_kernel<<<(MROWS * IDIMS / 4) / 256, 256, 0, stream>>>(x, T);
    packBt_kernel<<<(NOUT * KDIM) / 256, 256, 0, stream>>>(coeffs, Bt);

    dim3 grid(NOUT / BN, MROWS / BM);   // 8 x 64 workgroups
    lucas_wmma_gemm<<<grid, 256, 0, stream>>>(T, Bt, out);
}
